// MobilityGNN_29798483100345
// MI455X (gfx1250) — matrix operand order repaired and checked by reading, not compiled, not run on hardware
//
#include <hip/hip_runtime.h>
#include <cstdint>
#include <cstddef>

typedef __attribute__((ext_vector_type(16))) _Float16 v16h;
typedef __attribute__((ext_vector_type(8)))  float    v8f;

union AFrag { v16h v; _Float16 h[16]; };
union CFrag { v8f  v; float f[8]; };

// ---------------------------------------------------------------------------
// WMMA GEMM: Out[M, NT*16] = A[M,128] x W[128, NT*16]  (f32 in, f16 fragments,
// f32 accumulate). One wave computes a 16-row stripe across all NT col-tiles.
// W is staged to LDS *pre-swizzled into WMMA B-fragment order*, so each
// fragment load is one contiguous 32-byte LDS read (2x ds_load_b128).
//
// LDS layout (halves): [t][ks][lane][i], i contiguous:
//   Wlds[(((t*4+ks)*32 + lane)*16 + i)] = W[(ks*32 + (lane>>4)*16 + i)*NOUT
//                                           + t*16 + (lane&15)]
// ---------------------------------------------------------------------------
template<int NT>
__global__ __launch_bounds__(256) void gat_gemm_wmma(const float* __restrict__ A,
                                                     const float* __restrict__ W,
                                                     float* __restrict__ Out,
                                                     int M, int Mtiles) {
  constexpr int KD = 128;
  constexpr int NOUT = NT * 16;
  __shared__ __align__(32) _Float16 Wlds[KD * NOUT];

  const int tid = threadIdx.x;
  // swizzle-fill: idx -> (t, ks, lane, i)
  for (int idx = tid; idx < KD * NOUT; idx += 256) {
    const int i  = idx & 15;
    const int ln = (idx >> 4) & 31;
    const int ks = (idx >> 9) & 3;
    const int t  = idx >> 11;
    const int k   = ks * 32 + (ln >> 4) * 16 + i;
    const int col = t * 16 + (ln & 15);
    Wlds[idx] = (_Float16)W[k * NOUT + col];
  }
  __syncthreads();

  const int wave = tid >> 5;
  const int lane = tid & 31;
  const int rb = blockIdx.x * 8 + wave;
  if (rb >= Mtiles) return;

  const int hi = lane >> 4;           // half-wave id
  const int ln = lane & 15;
  int r = rb * 16 + ln;               // row this lane loads for A
  int r_ld = r < M ? r : (M - 1);     // clamp (M is a multiple of 16 here anyway)

  CFrag acc[NT];
  #pragma unroll
  for (int t = 0; t < NT; ++t)
    #pragma unroll
    for (int j = 0; j < 8; ++j) acc[t].f[j] = 0.f;

  const float* arow = A + (size_t)r_ld * KD;

  #pragma unroll
  for (int ks = 0; ks < KD / 32; ++ks) {
    // A fragment: lane<16 -> K {0..15}+32ks ; lane>=16 -> +16
    AFrag af;
    const int kb = ks * 32 + hi * 16;
    #pragma unroll
    for (int i = 0; i < 8; ++i) af.h[i]     = (_Float16)arow[kb + i];
    #pragma unroll
    for (int i = 0; i < 8; ++i) af.h[8 + i] = (_Float16)arow[kb + 8 + i];

    #pragma unroll
    for (int t = 0; t < NT; ++t) {
      // contiguous pre-swizzled B fragment: 32 bytes per lane
      AFrag bf;
      bf.v = *(const v16h*)&Wlds[(((t * 4 + ks) * 32) + lane) * 16];
      acc[t].v = __builtin_amdgcn_wmma_f32_16x16x32_f16(
          false, af.v, false, bf.v, (short)0, acc[t].v, false, false);
    }
  }

  #pragma unroll
  for (int t = 0; t < NT; ++t) {
    const int col = t * 16 + ln;
    #pragma unroll
    for (int j = 0; j < 8; ++j) {
      const int row = rb * 16 + j + hi * 8;   // D layout: VGPR j -> M = j (+8 for hi)
      if (row < M) Out[(size_t)row * NOUT + col] = acc[t].f[j];
    }
  }
}

// ---------------------------------------------------------------------------
// Per-(node,head) attention logits: a_s[n,h] = sum_c h[n,h,c]*att_s[h,c]
// Hh, C compile-time (powers of two) -> pure shift/mask indexing.
// ---------------------------------------------------------------------------
template<int Hh, int C>
__global__ void gat_logits(const float* __restrict__ Hm,
                           const float* __restrict__ aw_s,
                           const float* __restrict__ aw_d,
                           float* __restrict__ outs, float* __restrict__ outd,
                           int N) {
  int idx = blockIdx.x * blockDim.x + threadIdx.x;
  if (idx >= N * Hh) return;
  int n = idx / Hh, h = idx & (Hh - 1);
  const float* hp = Hm + (size_t)n * (Hh * C) + (size_t)h * C;
  float s = 0.f, d = 0.f;
  #pragma unroll 4
  for (int c = 0; c < C; ++c) {
    float v = hp[c];
    s = fmaf(v, aw_s[h * C + c], s);
    d = fmaf(v, aw_d[h * C + c], d);
  }
  outs[idx] = s;
  outd[idx] = d;
}

__global__ void gat_init_softmax(float* __restrict__ m, float* __restrict__ den, int n) {
  int i = blockIdx.x * blockDim.x + threadIdx.x;
  if (i < n) { m[i] = -1e30f; den[i] = 0.f; }
}

template<int F>
__global__ void gat_init_bias(float* __restrict__ out, const float* __restrict__ b, int N) {
  int i = blockIdx.x * blockDim.x + threadIdx.x;
  if (i < N * F) out[i] = b[i & (F - 1)];
}

__device__ inline float leaky02(float x) { return x > 0.f ? x : 0.2f * x; }

__device__ inline void atomic_max_f32(float* addr, float val) {
  int old = __float_as_int(*addr);
  while (__int_as_float(old) < val) {
    int assumed = old;
    old = atomicCAS((int*)addr, assumed, __float_as_int(val));
    if (old == assumed) break;
  }
}

// edges 0..E-1 come from edge_index; edges E..E+N-1 are self loops
__device__ inline void edge_ends(const int* ei, int E, int e, int& s, int& dn) {
  if (e < E) { s = ei[e]; dn = ei[E + e]; } else { s = dn = e - E; }
}

template<int Hh>
__global__ void gat_edge_max(const int* __restrict__ ei, int E, int N,
                             const float* __restrict__ as_, const float* __restrict__ ad_,
                             float* __restrict__ m) {
  int idx = blockIdx.x * blockDim.x + threadIdx.x;
  int tot = (E + N) * Hh;
  if (idx >= tot) return;
  int e = idx / Hh, h = idx & (Hh - 1);
  int s, dn; edge_ends(ei, E, e, s, dn);
  float el = leaky02(as_[s * Hh + h] + ad_[dn * Hh + h]);
  atomic_max_f32(&m[dn * Hh + h], el);
}

template<int Hh>
__global__ void gat_edge_sum(const int* __restrict__ ei, int E, int N,
                             const float* __restrict__ as_, const float* __restrict__ ad_,
                             const float* __restrict__ m, float* __restrict__ den) {
  int idx = blockIdx.x * blockDim.x + threadIdx.x;
  int tot = (E + N) * Hh;
  if (idx >= tot) return;
  int e = idx / Hh, h = idx & (Hh - 1);
  int s, dn; edge_ends(ei, E, e, s, dn);
  float el = leaky02(as_[s * Hh + h] + ad_[dn * Hh + h]);
  atomicAdd(&den[dn * Hh + h], __expf(el - m[dn * Hh + h]));
}

template<int Hh, int C>
__global__ void gat_edge_agg(const int* __restrict__ ei, int E, int N,
                             const float* __restrict__ as_, const float* __restrict__ ad_,
                             const float* __restrict__ m, const float* __restrict__ den,
                             const float* __restrict__ Hm, float* __restrict__ out) {
  constexpr int F = Hh * C;
  long long idx = (long long)blockIdx.x * blockDim.x + threadIdx.x;
  long long tot = (long long)(E + N) * F;
  if (idx >= tot) return;
  int e = (int)(idx / F);                 // F is a power of two -> shift
  int f = (int)idx & (F - 1);
  int h = f / C;                          // power of two -> shift
  int s, dn; edge_ends(ei, E, e, s, dn);
  float el = leaky02(as_[s * Hh + h] + ad_[dn * Hh + h]);
  float alpha = __expf(el - m[dn * Hh + h]) / den[dn * Hh + h];
  atomicAdd(&out[(size_t)dn * F + f], Hm[(size_t)s * F + f] * alpha);
}

__global__ void gat_elu(float* __restrict__ x, int n) {
  int i = blockIdx.x * blockDim.x + threadIdx.x;
  if (i < n) { float v = x[i]; x[i] = v > 0.f ? v : expm1f(v); }
}

// ---------------------------------------------------------------------------
extern "C" void kernel_launch(void* const* d_in, const int* in_sizes, int n_in,
                              void* d_out, int out_size, void* d_ws, size_t ws_size,
                              hipStream_t stream) {
  const float* x    = (const float*)d_in[0];
  const int*   ei   = (const int*)  d_in[1];   // [2,E] (src row, then dst row)
  const float* W1   = (const float*)d_in[2];
  const float* at_s1= (const float*)d_in[3];
  const float* at_d1= (const float*)d_in[4];
  const float* b1   = (const float*)d_in[5];
  const float* W2   = (const float*)d_in[6];
  const float* at_s2= (const float*)d_in[7];
  const float* at_d2= (const float*)d_in[8];
  const float* b2   = (const float*)d_in[9];
  float* dout = (float*)d_out;

  constexpr int Fin = 128, F1 = 128, H1 = 4, C1 = 32, F2 = 64, H2 = 1, C2 = 64;
  const int N  = in_sizes[0] / Fin;   // 50000
  const int E  = in_sizes[1] / 2;     // 800000
  const int Et = E + N;

  // workspace carve-out (floats)
  float* p  = (float*)d_ws;
  float* h1 = p; p += (size_t)N * F1;   // projected features, layer 1
  float* o1 = p; p += (size_t)N * F1;   // aggregated + ELU output of layer 1
  float* h2 = p; p += (size_t)N * F2;   // projected features, layer 2
  float* s1 = p; p += (size_t)N * H1;
  float* d1 = p; p += (size_t)N * H1;
  float* m1 = p; p += (size_t)N * H1;
  float* n1 = p; p += (size_t)N * H1;
  float* s2 = p; p += N;
  float* d2 = p; p += N;
  float* m2 = p; p += N;
  float* n2 = p; p += N;

  const int Mtiles = (N + 15) / 16;
  const dim3 blk(256);
  auto nb = [](long long tot, int b) { return (unsigned)((tot + b - 1) / b); };

  // ---------------- layer 1 ----------------
  gat_gemm_wmma<F1/16><<<dim3(nb(Mtiles, 8)), blk, 0, stream>>>(x, W1, h1, N, Mtiles);
  gat_logits<H1, C1><<<dim3(nb((long long)N * H1, 256)), blk, 0, stream>>>(h1, at_s1, at_d1, s1, d1, N);
  gat_init_softmax<<<dim3(nb((long long)N * H1, 256)), blk, 0, stream>>>(m1, n1, N * H1);
  gat_init_bias<F1><<<dim3(nb((long long)N * F1, 256)), blk, 0, stream>>>(o1, b1, N);
  gat_edge_max<H1><<<dim3(nb((long long)Et * H1, 256)), blk, 0, stream>>>(ei, E, N, s1, d1, m1);
  gat_edge_sum<H1><<<dim3(nb((long long)Et * H1, 256)), blk, 0, stream>>>(ei, E, N, s1, d1, m1, n1);
  gat_edge_agg<H1, C1><<<dim3(nb((long long)Et * F1, 256)), blk, 0, stream>>>(ei, E, N, s1, d1, m1, n1, h1, o1);
  gat_elu<<<dim3(nb((long long)N * F1, 256)), blk, 0, stream>>>(o1, N * F1);

  // ---------------- layer 2 ----------------
  gat_gemm_wmma<F2/16><<<dim3(nb(Mtiles, 8)), blk, 0, stream>>>(o1, W2, h2, N, Mtiles);
  gat_logits<H2, C2><<<dim3(nb((long long)N * H2, 256)), blk, 0, stream>>>(h2, at_s2, at_d2, s2, d2, N);
  gat_init_softmax<<<dim3(nb((long long)N * H2, 256)), blk, 0, stream>>>(m2, n2, N * H2);
  gat_init_bias<F2><<<dim3(nb((long long)N * F2, 256)), blk, 0, stream>>>(dout, b2, N);
  gat_edge_max<H2><<<dim3(nb((long long)Et * H2, 256)), blk, 0, stream>>>(ei, E, N, s2, d2, m2);
  gat_edge_sum<H2><<<dim3(nb((long long)Et * H2, 256)), blk, 0, stream>>>(ei, E, N, s2, d2, m2, n2);
  gat_edge_agg<H2, C2><<<dim3(nb((long long)Et * F2, 256)), blk, 0, stream>>>(ei, E, N, s2, d2, m2, n2, h2, dout);
  // concat=False with H=1: mean over heads is identity; bias already in dout.
}
